// VariantGAT_79843442032676
// MI455X (gfx1250) — matrix operand order repaired and checked by reading, not compiled, not run on hardware
//
#include <hip/hip_runtime.h>
#include <math.h>

typedef __attribute__((ext_vector_type(16))) __bf16 v16bf;
typedef __attribute__((ext_vector_type(8)))  __bf16 v8bf;
typedef __attribute__((ext_vector_type(8)))  float  v8f;

#define FIN 128
#define NEG_SLOPE 0.2f
#define GAT_EPS 1e-16f

// ---------------------------------------------------------------- fills
__global__ void fill_f32(float* p, float v, int n) {
    int i = blockIdx.x * blockDim.x + threadIdx.x;
    if (i < n) p[i] = v;
}
__global__ void fill_u32(unsigned* p, unsigned v, int n) {
    int i = blockIdx.x * blockDim.x + threadIdx.x;
    if (i < n) p[i] = v;
}

// ------------------------------------------------- float<->monotone key
__device__ __forceinline__ unsigned fkey(float f) {
    int i = __float_as_int(f);
    return (unsigned)(i ^ ((i >> 31) | 0x80000000));
}
__device__ __forceinline__ float fdecode(unsigned k) {
    int i = (int)((k & 0x80000000u) ? (k ^ 0x80000000u) : ~k);
    return __int_as_float(i);
}

__device__ __forceinline__ v8bf cvt8(float4 x, float4 y) {
    v8bf r;
    r[0] = (__bf16)x.x; r[1] = (__bf16)x.y; r[2] = (__bf16)x.z; r[3] = (__bf16)x.w;
    r[4] = (__bf16)y.x; r[5] = (__bf16)y.y; r[6] = (__bf16)y.z; r[7] = (__bf16)y.w;
    return r;
}

// ---------------------------------------------------------------- GEMM
// C[M,Nn] = A[M,K] x B[K,Nn], fp32 in/out, bf16 WMMA math.
// Tile BM=128, BN=64, BK=32; 256 threads = 8 waves.
// Wave w owns rows [16w,16w+16) x all 64 cols: 1 A-frag reused by 4 WMMAs.
// All B-fragments are preloaded so the 4 WMMAs can issue back-to-back.
__global__ __launch_bounds__(256)
void gemm_bf16_wmma(const float* __restrict__ A, const float* __restrict__ B,
                    float* __restrict__ C, int M, int K, int Nn) {
    // row strides padded to 40 halves = 80 bytes (16B-aligned, odd bank stride)
    __shared__ __align__(16) __bf16 As[128][40];   // [row][k]
    __shared__ __align__(16) __bf16 Bt[64][40];    // [col][k]  (B transposed)

    const int t    = threadIdx.x;
    const int wid  = t >> 5;         // 0..7 -> 16-row band
    const int lane = t & 31;
    const int g    = lane >> 4;      // lane half
    const int nl   = lane & 15;

    const int rowBase = blockIdx.x * 128;
    const int colBase = blockIdx.y * 64;

    // A staging map: 2 threads per row, 16 cols each
    const int ar = t >> 1;
    const int ac = (t & 1) * 16;
    const int arow  = rowBase + ar;
    const int crow  = arow < M ? arow : (M - 1);   // clamped, branch-free tail
    const bool avld = arow < M;

    // B staging map: 64 cols x 4 k-blocks of 8
    const int bc = t & 63;
    const int bk = (t >> 6) * 8;

    v8f acc[4] = {v8f{}, v8f{}, v8f{}, v8f{}};

    for (int k0 = 0; k0 < K; k0 += 32) {
        // ---- stage A 128x32: 4x global_load_b128 -> 16 bf16 -> 2x ds_store_b128
        {
            const float* ap = A + (long)crow * K + k0 + ac;
            float4 v0 = ((const float4*)ap)[0];
            float4 v1 = ((const float4*)ap)[1];
            float4 v2 = ((const float4*)ap)[2];
            float4 v3 = ((const float4*)ap)[3];
            if (!avld) {
                v0 = make_float4(0.f, 0.f, 0.f, 0.f); v1 = v0; v2 = v0; v3 = v0;
            }
            *(v8bf*)&As[ar][ac]     = cvt8(v0, v1);
            *(v8bf*)&As[ar][ac + 8] = cvt8(v2, v3);
        }
        // ---- stage B 32x64 transposed: 8 coalesced b32 loads -> ds_store_b128
        {
            const float* bp = B + (long)(k0 + bk) * Nn + colBase + bc;
            float4 u0, u1;
            u0.x = bp[0];        u0.y = bp[(long)Nn];   u0.z = bp[2L * Nn]; u0.w = bp[3L * Nn];
            u1.x = bp[4L * Nn];  u1.y = bp[5L * Nn];    u1.z = bp[6L * Nn]; u1.w = bp[7L * Nn];
            *(v8bf*)&Bt[bc][bk] = cvt8(u0, u1);
        }
        __syncthreads();

        // ---- A fragment: 16x32 per ISA 7.12.2 (lane-half holds K blocks of 8)
        const int arow_f = 16 * wid + nl;
        v8bf alo = *(const v8bf*)&As[arow_f][8 * g];
        v8bf ahi = *(const v8bf*)&As[arow_f][16 + 8 * g];
        v16bf a = __builtin_shufflevector(alo, ahi, 0, 1, 2, 3, 4, 5, 6, 7,
                                          8, 9, 10, 11, 12, 13, 14, 15);
        // ---- preload all 4 B fragments (B 32x16: lane-half g holds K blocks 8g..8g+7, 16+8g..16+8g+7)
        v16bf b[4];
        #pragma unroll
        for (int sub = 0; sub < 4; ++sub) {
            v8bf blo = *(const v8bf*)&Bt[16 * sub + nl][8 * g];
            v8bf bhi = *(const v8bf*)&Bt[16 * sub + nl][16 + 8 * g];
            b[sub] = __builtin_shufflevector(blo, bhi, 0, 1, 2, 3, 4, 5, 6, 7,
                                             8, 9, 10, 11, 12, 13, 14, 15);
        }
        // ---- 4 back-to-back WMMAs on the matrix pipe
        #pragma unroll
        for (int sub = 0; sub < 4; ++sub) {
            acc[sub] = __builtin_amdgcn_wmma_f32_16x16x32_bf16(
                false, a, false, b[sub], (short)0, acc[sub], false, false);
        }
        __syncthreads();
    }

    // ---- store: VGPR r -> row (r + 8*lanehalf), col = lane&15 within subtile
    #pragma unroll
    for (int sub = 0; sub < 4; ++sub) {
        #pragma unroll
        for (int r = 0; r < 8; ++r) {
            int row = rowBase + 16 * wid + r + 8 * g;
            if (row < M) C[(long)row * Nn + colBase + 16 * sub + nl] = acc[sub][r];
        }
    }
}

// ------------------------------------------- per-(node,head) attn dots
__global__ __launch_bounds__(256)
void attn_logits(const float* __restrict__ h, const float* __restrict__ asrc,
                 const float* __restrict__ adst, float* __restrict__ als,
                 float* __restrict__ ald, int N, int H, int C) {
    int wid  = (int)((blockIdx.x * (unsigned)blockDim.x + threadIdx.x) >> 5);
    int lane = threadIdx.x & 31;
    if (wid >= N * H) return;
    int n = wid / H, hh = wid % H;
    const float* hp = h + (long)n * H * C + (long)hh * C;
    const float* ap = asrc + hh * C;
    const float* bp = adst + hh * C;
    float s = 0.0f, d = 0.0f;
    for (int c = lane; c < C; c += 32) {
        float v = hp[c];
        s += v * ap[c];
        d += v * bp[c];
    }
    #pragma unroll
    for (int o = 16; o > 0; o >>= 1) {
        s += __shfl_xor(s, o, 32);
        d += __shfl_xor(d, o, 32);
    }
    if (lane == 0) { als[wid] = s; ald[wid] = d; }
}

// --------------------------------------------------- edge segment max
__global__ void edge_max(const long long* __restrict__ e0, const long long* __restrict__ e1,
                         int E, int N, int H, const float* __restrict__ als,
                         const float* __restrict__ ald, unsigned* __restrict__ mkey) {
    int e = blockIdx.x * blockDim.x + threadIdx.x;
    if (e >= E + N) return;
    int s, d;
    if (e < E) { s = (int)e0[e]; d = (int)e1[e]; } else { s = d = e - E; }
    for (int h = 0; h < H; ++h) {
        float v = als[s * H + h] + ald[d * H + h];
        v = v > 0.0f ? v : NEG_SLOPE * v;
        atomicMax(&mkey[d * H + h], fkey(v));
    }
}

__global__ void decode_max(const unsigned* __restrict__ mkey, float* __restrict__ mval, int n) {
    int i = blockIdx.x * blockDim.x + threadIdx.x;
    if (i < n) mval[i] = fdecode(mkey[i]);
}

// ------------------------------------------- edge exp + denom (segsum)
__global__ void edge_expsum(const long long* __restrict__ e0, const long long* __restrict__ e1,
                            int E, int N, int H, const float* __restrict__ als,
                            const float* __restrict__ ald, const float* __restrict__ mval,
                            float* __restrict__ exbuf, float* __restrict__ denom) {
    int e = blockIdx.x * blockDim.x + threadIdx.x;
    if (e >= E + N) return;
    int s, d;
    if (e < E) { s = (int)e0[e]; d = (int)e1[e]; } else { s = d = e - E; }
    for (int h = 0; h < H; ++h) {
        float v = als[s * H + h] + ald[d * H + h];
        v = v > 0.0f ? v : NEG_SLOPE * v;
        float ex = expf(v - mval[d * H + h]);
        exbuf[(long)e * H + h] = ex;
        atomicAdd(&denom[d * H + h], ex);
    }
}

// -------------------------------------- wave-per-edge weighted scatter
__global__ __launch_bounds__(256)
void edge_scatter(const long long* __restrict__ e0, const long long* __restrict__ e1,
                  int E, int N, int H, int C, const float* __restrict__ hlin,
                  const float* __restrict__ exbuf, const float* __restrict__ denom,
                  float* __restrict__ acc) {
    int wid  = (int)((blockIdx.x * (unsigned)blockDim.x + threadIdx.x) >> 5);
    int lane = threadIdx.x & 31;
    if (wid >= E + N) return;
    int s, d;
    if (wid < E) { s = (int)e0[wid]; d = (int)e1[wid]; } else { s = d = wid - E; }
    int HC = H * C;
    const float* hs  = hlin  + (long)s * HC;
    float*       ad  = acc   + (long)d * HC;
    const float* exr = exbuf + (long)wid * H;
    const float* der = denom + (long)d * H;
    for (int f = lane; f < HC; f += 32) {
        int hh = f / C;
        float alpha = exr[hh] / (der[hh] + GAT_EPS);
        atomicAdd(&ad[f], alpha * hs[f]);
    }
}

// ------------------------------------------------- bias (+ELU) epilog
__global__ void finalize(const float* __restrict__ acc, const float* __restrict__ bias,
                         float* __restrict__ out, int total, int HC, int do_elu) {
    int i = blockIdx.x * blockDim.x + threadIdx.x;
    if (i >= total) return;
    float v = acc[i] + bias[i % HC];
    if (do_elu) v = v > 0.0f ? v : (expf(v) - 1.0f);
    out[i] = v;
}

// ------------------------------------------------ classifier, wave/row
__global__ __launch_bounds__(32)
void classifier(const float* __restrict__ h3, const long long* __restrict__ gidx,
                const float* __restrict__ cW1, const float* __restrict__ cb1,
                const float* __restrict__ cW2, const float* __restrict__ cb2,
                float* __restrict__ out, int B, int F) {
    int b = blockIdx.x;
    if (b >= B) return;
    int j = threadIdx.x;               // hidden unit 0..31
    const float* f = h3 + (long)gidx[b] * F;
    float hid = cb1[j];
    for (int k = 0; k < F; ++k) hid += f[k] * cW1[k * 32 + j];
    hid = hid > 0.0f ? hid : 0.0f;
    #pragma unroll
    for (int o = 0; o < 2; ++o) {
        float p = hid * cW2[j * 2 + o];
        #pragma unroll
        for (int q = 16; q > 0; q >>= 1) p += __shfl_xor(p, q, 32);
        if (j == 0) out[b * 2 + o] = p + cb2[o];
    }
}

extern "C" void kernel_launch(void* const* d_in, const int* in_sizes, int n_in,
                              void* d_out, int out_size, void* d_ws, size_t ws_size,
                              hipStream_t stream) {
    const float*     x     = (const float*)d_in[0];
    const long long* ei    = (const long long*)d_in[1];
    const long long* gidx  = (const long long*)d_in[2];
    const float* W1 = (const float*)d_in[3];
    const float* as1 = (const float*)d_in[4];
    const float* ad1 = (const float*)d_in[5];
    const float* b1 = (const float*)d_in[6];
    const float* W2 = (const float*)d_in[7];
    const float* as2 = (const float*)d_in[8];
    const float* ad2 = (const float*)d_in[9];
    const float* b2 = (const float*)d_in[10];
    const float* W3 = (const float*)d_in[11];
    const float* as3 = (const float*)d_in[12];
    const float* ad3 = (const float*)d_in[13];
    const float* b3 = (const float*)d_in[14];
    const float* cW1 = (const float*)d_in[15];
    const float* cb1 = (const float*)d_in[16];
    const float* cW2 = (const float*)d_in[17];
    const float* cb2 = (const float*)d_in[18];

    const int N    = in_sizes[0] / FIN;      // 50000
    const int E    = in_sizes[1] / 2;        // 400000
    const int B    = in_sizes[2];            // 1024
    const int Etot = E + N;                  // self-loops appended
    const long long* src = ei;
    const long long* dst = ei + E;

    // ---- workspace carve (floats)
    float* ws = (float*)d_ws;
    size_t o = 0;
    float*    bufA  = ws + o; o += (size_t)N * 512;   // hlin (layers 1/2), L3 scratch
    float*    bufB  = ws + o; o += (size_t)N * 512;   // acc  (layers 1/2)
    float*    bufC  = ws + o; o += (size_t)N * 512;   // layer outputs 1/2
    float*    als   = ws + o; o += (size_t)N * 8;
    float*    ald   = ws + o; o += (size_t)N * 8;
    unsigned* mkey  = (unsigned*)(ws + o); o += (size_t)N * 8;
    float*    mval  = ws + o; o += (size_t)N * 8;
    float*    denom = ws + o; o += (size_t)N * 8;
    float*    exbuf = ws + o; o += (size_t)Etot * 8;
    (void)ws_size; (void)n_in; (void)out_size;

    auto run_layer = [&](const float* xin, int K, const float* W, const float* av,
                         const float* bv, const float* bias, int H, int C,
                         float* hlin, float* acc, float* xout, int do_elu) {
        const int Nn = H * C;
        // 1) projection GEMM (bf16 WMMA)
        dim3 gg((N + 127) / 128, Nn / 64);
        gemm_bf16_wmma<<<gg, 256, 0, stream>>>(xin, W, hlin, N, K, Nn);
        // 2) per-(node,head) attention dots
        int nh = N * H;
        attn_logits<<<(nh + 7) / 8, 256, 0, stream>>>(hlin, av, bv, als, ald, N, H, C);
        // 3) init reductions
        fill_u32<<<(nh + 255) / 256, 256, 0, stream>>>(mkey, 0u, nh);
        fill_f32<<<(nh + 255) / 256, 256, 0, stream>>>(denom, 0.0f, nh);
        fill_f32<<<((N * Nn) + 255) / 256, 256, 0, stream>>>(acc, 0.0f, N * Nn);
        // 4) segment max over dst
        edge_max<<<(Etot + 255) / 256, 256, 0, stream>>>(src, dst, E, N, H, als, ald, mkey);
        decode_max<<<(nh + 255) / 256, 256, 0, stream>>>(mkey, mval, nh);
        // 5) exp + segment sum
        edge_expsum<<<(Etot + 255) / 256, 256, 0, stream>>>(src, dst, E, N, H, als, ald,
                                                            mval, exbuf, denom);
        // 6) alpha-weighted scatter (wave per edge)
        edge_scatter<<<(Etot + 7) / 8, 256, 0, stream>>>(src, dst, E, N, H, C, hlin,
                                                         exbuf, denom, acc);
        // 7) bias + optional ELU
        finalize<<<((N * Nn) + 255) / 256, 256, 0, stream>>>(acc, bias, xout, N * Nn, Nn, do_elu);
    };

    // layer 1: Fin=128 -> 8x64, concat, ELU
    run_layer(x,    FIN, W1, as1, ad1, b1, 8, 64, bufA, bufB, bufC, 1);
    // layer 2: 512 -> 8x64, concat, ELU
    run_layer(bufC, 512, W2, as2, ad2, b2, 8, 64, bufA, bufB, bufC, 1);
    // layer 3: 512 -> 1x64, mean(H=1) == identity, no ELU; carve from freed bufA
    float* h3lin = bufA;
    float* acc3  = bufA + (size_t)N * 64;
    float* h3    = bufA + (size_t)2 * N * 64;
    run_layer(bufC, 512, W3, as3, ad3, b3, 1, 64, h3lin, acc3, h3, 0);

    // classifier head -> d_out [B,2]
    classifier<<<B, 32, 0, stream>>>(h3, gidx, cW1, cb1, cW2, cb2, (float*)d_out, B, 64);
}
